// HyperVigClassifier_40389872452042
// MI455X (gfx1250) — matrix operand order repaired and checked by reading, not compiled, not run on hardware
//
#include <hip/hip_runtime.h>
#include <hip/hip_bf16.h>
#include <math.h>

// ---------------------------------------------------------------------------
// Types for CDNA5 WMMA (wave32): v_wmma_f32_16x16x32_bf16
// ---------------------------------------------------------------------------
typedef __bf16 bf16_t;
typedef __attribute__((ext_vector_type(16))) __bf16 v16bf;
typedef __attribute__((ext_vector_type(8)))  __bf16 v8bf;
typedef __attribute__((ext_vector_type(8)))  float  v8f;

#define K_NODES   65536
#define K_EDGES   65536
#define K_GRAPHS  256
#define K_HID     512
#define K_INCH    384
#define K_CLS     1000
#define K_CLSPAD  1024

// ---------------------------------------------------------------------------
// Fragment loaders.
// A (16x32 bf16, MxK): lane L -> row M = L&15 ; element e -> K = (L>>4)*16 + e.
// B (32x16 bf16, KxN): lane L -> col N = L&15 ; element e -> K = (L>>4)*16 + e.
// Both read from row-major [row, K] storage (A = activations, B = W^T).
// ---------------------------------------------------------------------------
__device__ __forceinline__ v16bf load_frag_a(const bf16_t* __restrict__ rowk, int half8) {
  v8bf lo = *(const v8bf*)(rowk + 2 * half8);
  v8bf hi = *(const v8bf*)(rowk + 2 * half8 + 8);
  return __builtin_shufflevector(lo, hi, 0,1,2,3,4,5,6,7,8,9,10,11,12,13,14,15);
}
__device__ __forceinline__ v16bf load_frag_b(const bf16_t* __restrict__ rowk, int half16) {
  const bf16_t* q = rowk + half16;
  v8bf lo = *(const v8bf*)(q);
  v8bf hi = *(const v8bf*)(q + 8);
  return __builtin_shufflevector(lo, hi, 0,1,2,3,4,5,6,7,8,9,10,11,12,13,14,15);
}

// ---------------------------------------------------------------------------
// bf16 WMMA GEMM: C[M,ldc](f32) = A[M,K](bf16) * Bt[Npad,K](bf16)^T (+bias)(+relu)
// Block = 256 threads = 8 waves; block tile 128(M) x 64(N); wave tile 16 x 64.
// M % 128 == 0, K % 32 == 0 required. N guarded at store by Nout.
// ---------------------------------------------------------------------------
__global__ __launch_bounds__(256)
void gemm_bf16_wmma(const bf16_t* __restrict__ A, const bf16_t* __restrict__ Bt,
                    float* __restrict__ C, const float* __restrict__ bias,
                    int M, int Nout, int K, int ldc, int relu)
{
  const int lane = threadIdx.x & 31;
  const int wave = threadIdx.x >> 5;
  const int m0   = blockIdx.y * 128 + wave * 16;
  const int n0   = blockIdx.x * 64;
  const int r    = lane & 15;
  const int h8   = (lane >> 4) * 8;
  const int h16  = (lane >> 4) * 16;

  const bf16_t* arow = A  + (size_t)(m0 + r) * K;
  const bf16_t* brow = Bt + (size_t)(n0 + r) * K;
  const size_t bstep = (size_t)16 * K;

  v8f acc[4] = { v8f{}, v8f{}, v8f{}, v8f{} };

  for (int k = 0; k < K; k += 32) {
    v16bf a  = load_frag_a(arow + k, h8);
    v16bf b0 = load_frag_b(brow + k,             h16);
    v16bf b1 = load_frag_b(brow + bstep + k,     h16);
    v16bf b2 = load_frag_b(brow + 2 * bstep + k, h16);
    v16bf b3 = load_frag_b(brow + 3 * bstep + k, h16);
    acc[0] = __builtin_amdgcn_wmma_f32_16x16x32_bf16(false, a, false, b0, (short)0, acc[0], false, false);
    acc[1] = __builtin_amdgcn_wmma_f32_16x16x32_bf16(false, a, false, b1, (short)0, acc[1], false, false);
    acc[2] = __builtin_amdgcn_wmma_f32_16x16x32_bf16(false, a, false, b2, (short)0, acc[2], false, false);
    acc[3] = __builtin_amdgcn_wmma_f32_16x16x32_bf16(false, a, false, b3, (short)0, acc[3], false, false);
  }

  // C/D layout: element g of lane L -> M = m0 + g + 8*(L>>4), N = n0 + 16*j + (L&15)
  const int mbase = m0 + (lane >> 4) * 8;
  #pragma unroll
  for (int g = 0; g < 8; ++g) {
    float* crow = C + (size_t)(mbase + g) * ldc;
    #pragma unroll
    for (int j = 0; j < 4; ++j) {
      int n = n0 + j * 16 + r;
      if (n < Nout) {
        float v = acc[j][g] + (bias ? bias[n] : 0.0f);
        if (relu) v = fmaxf(v, 0.0f);
        crow[n] = v;
      }
    }
  }
}

// ---------------------------------------------------------------------------
// Elementwise / sparse helper kernels
// ---------------------------------------------------------------------------
__global__ void zero_b128_kernel(uint4* __restrict__ p, size_t n16) {
  size_t i = (size_t)blockIdx.x * blockDim.x + threadIdx.x;
  if (i < n16) p[i] = uint4{0u, 0u, 0u, 0u};
}

__global__ void cvt_f32_bf16_kernel(const float* __restrict__ in, bf16_t* __restrict__ out, size_t n) {
  size_t i = (size_t)blockIdx.x * blockDim.x + threadIdx.x;
  if (i < n) out[i] = (bf16_t)in[i];
}

// W [K,N] f32  ->  Wt [Npad,K] bf16 (rows >= N zero-filled)
__global__ void transpose_bf16_kernel(const float* __restrict__ W, bf16_t* __restrict__ Wt,
                                      int K, int N, int Npad) {
  size_t i = (size_t)blockIdx.x * blockDim.x + threadIdx.x;
  size_t total = (size_t)Npad * K;
  if (i < total) {
    int n = (int)(i / K);
    int k = (int)(i % K);
    Wt[i] = (n < N) ? (bf16_t)W[(size_t)k * N + n] : (bf16_t)0.0f;
  }
}

__global__ void count_deg_kernel(const int* __restrict__ row, const int* __restrict__ col,
                                 float* __restrict__ D, float* __restrict__ B, int nnz) {
  int i = blockIdx.x * blockDim.x + threadIdx.x;
  if (i < nnz) {
    atomicAdd(&D[row[i]], 1.0f);
    atomicAdd(&B[col[i]], 1.0f);
  }
}

__global__ void invert_kernel(float* __restrict__ v, int n) {
  int i = blockIdx.x * blockDim.x + threadIdx.x;
  if (i < n) { float d = v[i]; v[i] = (d > 0.0f) ? 1.0f / d : 0.0f; }
}

// dst[didx[e], :] += scale[sidx[e]] * src[sidx[e], :]   (C = 512)
// Two edges per 256-thread block; each thread owns 4 channels (float4 gather).
__global__ __launch_bounds__(256)
void scatter_add_kernel(const float* __restrict__ src, const float* __restrict__ scale,
                        const int* __restrict__ sidx, const int* __restrict__ didx,
                        float* __restrict__ dst, int nnz) {
  int e = blockIdx.x * 2 + (threadIdx.x >> 7);
  if (e >= nnz) return;
  int t = threadIdx.x & 127;
  int s = sidx[e];
  int d = didx[e];
  float sc = scale ? scale[s] : 1.0f;
  const float4 v = ((const float4*)(src + (size_t)s * 512))[t];
  float* dp = dst + (size_t)d * 512 + (size_t)t * 4;
  atomicAdd(dp + 0, v.x * sc);
  atomicAdd(dp + 1, v.y * sc);
  atomicAdd(dp + 2, v.z * sc);
  atomicAdd(dp + 3, v.w * sc);
}

// act = bf16(relu(acc * Dinv[row] + bias[ch]))
__global__ void finish_layer_kernel(const float* __restrict__ acc, const float* __restrict__ dinv,
                                    const float* __restrict__ bias, bf16_t* __restrict__ act,
                                    size_t total) {
  size_t i = (size_t)blockIdx.x * blockDim.x + threadIdx.x;
  if (i < total) {
    float v = acc[i] * dinv[i >> 9] + bias[i & 511];
    act[i] = (bf16_t)fmaxf(v, 0.0f);
  }
}

// gate[row] = dot(g1[row,:512], w) + b  — one wave per row
__global__ void gate_dot_kernel(const float* __restrict__ g1, const float* __restrict__ w,
                                const float* __restrict__ b, float* __restrict__ gate, int n) {
  int rowi = blockIdx.x * 8 + (threadIdx.x >> 5);
  int lane = threadIdx.x & 31;
  float s = 0.0f;
  const float* rp = g1 + (size_t)rowi * 512;
  for (int c = lane; c < 512; c += 32) s += rp[c] * w[c];
  for (int off = 16; off; off >>= 1) s += __shfl_xor(s, off, 32);
  if (lane == 0 && rowi < n) gate[rowi] = s + b[0];
}

// order-preserving float <-> uint for atomicMax
__device__ __forceinline__ unsigned f2key(float f) {
  unsigned u = __float_as_uint(f);
  return (u & 0x80000000u) ? ~u : (u | 0x80000000u);
}
__device__ __forceinline__ float key2f(unsigned k) {
  unsigned u = (k & 0x80000000u) ? (k & 0x7FFFFFFFu) : ~k;
  return __uint_as_float(u);
}

__global__ void seg_max_kernel(const float* __restrict__ gate, const int* __restrict__ bm,
                               unsigned* __restrict__ mkey, int n) {
  int i = blockIdx.x * blockDim.x + threadIdx.x;
  if (i < n) atomicMax(&mkey[bm[i]], f2key(gate[i]));
}

__global__ void seg_exp_kernel(const float* __restrict__ gate, const int* __restrict__ bm,
                               const unsigned* __restrict__ mkey, float* __restrict__ e,
                               float* __restrict__ z, int n) {
  int i = blockIdx.x * blockDim.x + threadIdx.x;
  if (i < n) {
    int g = bm[i];
    float ev = __expf(gate[i] - key2f(mkey[g]));
    e[i] = ev;
    atomicAdd(&z[g], ev);
  }
}

// pooled[g,:] += (e[n]/z[g]) * h[n,:]  — one node per block, 128 thr x 4ch
__global__ void pool_kernel(const bf16_t* __restrict__ act, const float* __restrict__ e,
                            const float* __restrict__ z, const int* __restrict__ bm,
                            float* __restrict__ pooled) {
  int nIdx = blockIdx.x;
  int g = bm[nIdx];
  float alpha = e[nIdx] / z[g];
  int t = threadIdx.x;
  const bf16_t* ap = act + (size_t)nIdx * 512 + (size_t)t * 4;
  float* pp = pooled + (size_t)g * 512 + (size_t)t * 4;
  #pragma unroll
  for (int j = 0; j < 4; ++j) atomicAdd(pp + j, alpha * (float)ap[j]);
}

// ---------------------------------------------------------------------------
// Host launcher
// ---------------------------------------------------------------------------
extern "C" void kernel_launch(void* const* d_in, const int* in_sizes, int n_in,
                              void* d_out, int out_size, void* d_ws, size_t ws_size,
                              hipStream_t stream) {
  (void)n_in; (void)out_size; (void)ws_size;

  const float* x   = (const float*)d_in[0];
  const int*   ei  = (const int*)d_in[1];
  const int*   bm  = (const int*)d_in[2];
  const float* W1  = (const float*)d_in[3];  const float* b1  = (const float*)d_in[4];
  const float* W2  = (const float*)d_in[5];  const float* b2  = (const float*)d_in[6];
  const float* W3  = (const float*)d_in[7];  const float* b3  = (const float*)d_in[8];
  const float* gW1 = (const float*)d_in[9];  const float* gb1 = (const float*)d_in[10];
  const float* gW2 = (const float*)d_in[11]; const float* gb2 = (const float*)d_in[12];
  const float* cW  = (const float*)d_in[13]; const float* cb  = (const float*)d_in[14];

  const int nnz = in_sizes[1] / 2;
  const int* row = ei;
  const int* col = ei + nnz;

  // workspace carve-out (256B aligned regions)
  char* ws = (char*)d_ws;
  size_t off = 0;
  auto carve = [&](size_t bytes) -> char* {
    char* p = ws + off;
    off += (bytes + 255) & ~(size_t)255;
    return p;
  };
  bf16_t*  act    = (bf16_t*)carve((size_t)K_NODES * K_HID * 2);   // GEMM input activations
  float*   bufY   = (float*) carve((size_t)K_NODES * K_HID * 4);   // GEMM out / node accumulator
  float*   bufHE  = (float*) carve((size_t)K_EDGES * K_HID * 4);   // hyperedge accumulator / g1
  bf16_t*  W1t    = (bf16_t*)carve((size_t)K_HID * K_INCH * 2);
  bf16_t*  W2t    = (bf16_t*)carve((size_t)K_HID * K_HID * 2);
  bf16_t*  W3t    = (bf16_t*)carve((size_t)K_HID * K_HID * 2);
  bf16_t*  gW1t   = (bf16_t*)carve((size_t)K_HID * K_HID * 2);
  bf16_t*  cWt    = (bf16_t*)carve((size_t)K_CLSPAD * K_HID * 2);
  float*   Dinv   = (float*) carve((size_t)K_NODES * 4);
  float*   Binv   = (float*) carve((size_t)K_EDGES * 4);
  float*   gate   = (float*) carve((size_t)K_NODES * 4);
  float*   ebuf   = (float*) carve((size_t)K_NODES * 4);
  unsigned* mkey  = (unsigned*)carve((size_t)K_GRAPHS * 4);
  float*   zsum   = (float*) carve((size_t)K_GRAPHS * 4);
  float*   pooled = (float*) carve((size_t)K_GRAPHS * K_HID * 4);
  bf16_t*  pooledb= (bf16_t*)carve((size_t)K_GRAPHS * K_HID * 2);

  auto blocks = [](size_t n) { return (unsigned)((n + 255) / 256); };
  // n_u32 must be a multiple of 4 (all our buffers are)
  auto zero_f = [&](void* p, size_t n_u32) {
    size_t n16 = n_u32 / 4;
    zero_b128_kernel<<<blocks(n16), 256, 0, stream>>>((uint4*)p, n16);
  };

  // ---- prep: weights -> bf16 W^T, x -> bf16, degrees -------------------
  transpose_bf16_kernel<<<blocks((size_t)K_HID * K_INCH), 256, 0, stream>>>(W1, W1t, K_INCH, K_HID, K_HID);
  transpose_bf16_kernel<<<blocks((size_t)K_HID * K_HID), 256, 0, stream>>>(W2, W2t, K_HID, K_HID, K_HID);
  transpose_bf16_kernel<<<blocks((size_t)K_HID * K_HID), 256, 0, stream>>>(W3, W3t, K_HID, K_HID, K_HID);
  transpose_bf16_kernel<<<blocks((size_t)K_HID * K_HID), 256, 0, stream>>>(gW1, gW1t, K_HID, K_HID, K_HID);
  transpose_bf16_kernel<<<blocks((size_t)K_CLSPAD * K_HID), 256, 0, stream>>>(cW, cWt, K_HID, K_CLS, K_CLSPAD);
  cvt_f32_bf16_kernel<<<blocks((size_t)K_NODES * K_INCH), 256, 0, stream>>>(x, act, (size_t)K_NODES * K_INCH);

  zero_f(Dinv, K_NODES);
  zero_f(Binv, K_EDGES);
  count_deg_kernel<<<blocks(nnz), 256, 0, stream>>>(row, col, Dinv, Binv, nnz);
  invert_kernel<<<blocks(K_NODES), 256, 0, stream>>>(Dinv, K_NODES);
  invert_kernel<<<blocks(K_EDGES), 256, 0, stream>>>(Binv, K_EDGES);

  const size_t featN = (size_t)K_NODES * K_HID;
  const size_t featE = (size_t)K_EDGES * K_HID;
  const unsigned sblocks = (unsigned)((nnz + 1) / 2);

  // ---- hconv layer: act = relu(Dinv * S_row( Binv * S_col( act @ W ) ) + b)
  auto hconv = [&](const bf16_t* Wt, const float* bias, int K) {
    dim3 grid(K_HID / 64, K_NODES / 128);
    gemm_bf16_wmma<<<grid, 256, 0, stream>>>(act, Wt, bufY, nullptr, K_NODES, K_HID, K, K_HID, 0);
    zero_f(bufHE, featE);
    // nodes -> hyperedges (unscaled)
    scatter_add_kernel<<<sblocks, 256, 0, stream>>>(bufY, nullptr, row, col, bufHE, nnz);
    zero_f(bufY, featN);
    // hyperedges -> nodes, with B^{-1} fused into the gather
    scatter_add_kernel<<<sblocks, 256, 0, stream>>>(bufHE, Binv, col, row, bufY, nnz);
    finish_layer_kernel<<<blocks(featN), 256, 0, stream>>>(bufY, Dinv, bias, act, featN);
  };
  hconv(W1t, b1, K_INCH);
  hconv(W2t, b2, K_HID);
  hconv(W3t, b3, K_HID);

  // ---- attentional aggregation ----------------------------------------
  {
    dim3 grid(K_HID / 64, K_NODES / 128);  // g1 = relu(h @ gW1 + gb1) into bufHE (f32)
    gemm_bf16_wmma<<<grid, 256, 0, stream>>>(act, gW1t, bufHE, gb1, K_NODES, K_HID, K_HID, K_HID, 1);
  }
  gate_dot_kernel<<<K_NODES / 8, 256, 0, stream>>>(bufHE, gW2, gb2, gate, K_NODES);

  zero_f(mkey, K_GRAPHS);
  zero_f(zsum, K_GRAPHS);
  seg_max_kernel<<<blocks(K_NODES), 256, 0, stream>>>(gate, bm, mkey, K_NODES);
  seg_exp_kernel<<<blocks(K_NODES), 256, 0, stream>>>(gate, bm, mkey, ebuf, zsum, K_NODES);
  zero_f(pooled, (size_t)K_GRAPHS * K_HID);
  pool_kernel<<<K_NODES, 128, 0, stream>>>(act, ebuf, zsum, bm, pooled);

  // ---- classifier: out[256,1000] = pooled @ cW + cb (WMMA, padded N) ---
  cvt_f32_bf16_kernel<<<blocks((size_t)K_GRAPHS * K_HID), 256, 0, stream>>>(
      pooled, pooledb, (size_t)K_GRAPHS * K_HID);
  {
    dim3 grid(K_CLSPAD / 64, K_GRAPHS / 128);
    gemm_bf16_wmma<<<grid, 256, 0, stream>>>(pooledb, cWt, (float*)d_out, cb,
                                             K_GRAPHS, K_CLS, K_HID, K_CLS, 0);
  }
}
